// CausalSelfAttention_73005854097696
// MI455X (gfx1250) — matrix operand order repaired and checked by reading, not compiled, not run on hardware
//
#include <hip/hip_runtime.h>
#include <hip/hip_bf16.h>
#include <math.h>

typedef _Float16 f16;
typedef __attribute__((ext_vector_type(16))) _Float16 v16h;
typedef __attribute__((ext_vector_type(8)))  _Float16 v8h;
typedef __attribute__((ext_vector_type(8)))  float    v8f;
typedef __attribute__((ext_vector_type(4)))  int      v4i;

union AF { v16h v; v8h h[2]; };

constexpr int Bsz = 4, L = 2048, D = 1024, H = 16, HD = 64;
constexpr int M = Bsz * L;                 // 8192 rows
constexpr float SCALE = 0.125f;            // 1/sqrt(64)

// workspace layout (bytes)
constexpr size_t OFF_QKV = 0;                                    // f16 [M][3D]
constexpr size_t OFF_Q   = OFF_QKV + (size_t)M * 3 * D * 2;      // f16 [B][H][L][HD]
constexpr size_t OFF_K   = OFF_Q  + (size_t)Bsz * H * L * HD * 2;
constexpr size_t OFF_V   = OFF_K  + (size_t)Bsz * H * L * HD * 2;
constexpr size_t OFF_AT  = OFF_V  + (size_t)Bsz * H * L * HD * 2; // f16 [M][D]

// ---------------------------------------------------------------------------
// gfx1250 async global->LDS copy (GLOBAL_LOAD_ASYNC_TO_LDS_B128, ASYNCcnt).
// Builtin parameter type (from hipcc diagnostic): pointer to int4 in the
// global address space; LDS side is the 32-bit AS(3) pointer.
// Guarded: falls back to b128 load + ds_store when the builtin is absent.
// ---------------------------------------------------------------------------
#if defined(__has_builtin)
#  if __has_builtin(__builtin_amdgcn_global_load_async_to_lds_b128) && \
      __has_builtin(__builtin_amdgcn_s_wait_asynccnt)
#    define USE_ASYNC_LDS 1
#  endif
#endif
#ifndef USE_ASYNC_LDS
#  define USE_ASYNC_LDS 0
#endif

#if USE_ASYNC_LDS
typedef __attribute__((address_space(1))) v4i* g128p;
typedef __attribute__((address_space(3))) v4i* l128p;
__device__ __forceinline__ void cp16_g2l(const void* g, void* l) {
  __builtin_amdgcn_global_load_async_to_lds_b128(
      (g128p)(unsigned long long)(uintptr_t)g,
      (l128p)(unsigned)(uintptr_t)l,   // low 32 bits of generic LDS addr = offset
      0, 0);
}
__device__ __forceinline__ void cp_wait() { __builtin_amdgcn_s_wait_asynccnt(0); }
#else
__device__ __forceinline__ void cp16_g2l(const void* g, void* l) {
  *(v8h*)l = *(const v8h*)g;
}
__device__ __forceinline__ void cp_wait() {}
#endif

__device__ __forceinline__ v8f wmma16(v16h a, v16h b, v8f c) {
  // D = A(16x32 f16) * B(32x16 f16) + C(16x16 f32)
  return __builtin_amdgcn_wmma_f32_16x16x32_f16(false, a, false, b,
                                                (short)0, c, false, false);
}

// ---------------------------------------------------------------------------
// Tiled GEMM: Out[M x Nd] = A[M x Kd] * W[Kd x Nd] + bias.
// 256 threads = 8 waves; 128x64 block tile; each wave owns a 32x32 register
// tile (2 A-frags x 2 B-frags -> 4 WMMAs per 32-K step).
// ---------------------------------------------------------------------------
template <bool A_F32, bool OUT_F32>
__global__ __launch_bounds__(256) void gemm_tile128(const void* __restrict__ Ap,
                                                    const float* __restrict__ W,
                                                    const float* __restrict__ bias,
                                                    void* __restrict__ Outp,
                                                    int Kd, int Nd) {
  __shared__ alignas(16) f16 As[128 * 32];  // [row][k], stride 32
  __shared__ alignas(16) f16 Bs[64 * 32];   // transposed: [col][k], stride 32

  const float* Af = (const float*)Ap;
  const f16*   Ah = (const f16*)Ap;
  float*       Of = (float*)Outp;
  f16*         Oh = (f16*)Outp;

  const int tid  = threadIdx.x;
  const int lane = tid & 31;
  const int wv   = tid >> 5;
  const int mw   = wv & 3;       // wave row group (0..3) -> 32 rows each
  const int nw   = wv >> 2;      // wave col group (0..1) -> 32 cols each
  const int m0   = blockIdx.x * 128;
  const int n0   = blockIdx.y * 64;

  const int ma   = lane & 15;
  const int lh   = lane >> 4;
  const int k0l  = lh * 8;       // A-frag K start within 32-window
  const int kb16 = lh * 16;      // unused; B-frags also start at k0l

  v8f acc[2][2];
#pragma unroll
  for (int jm = 0; jm < 2; ++jm)
#pragma unroll
    for (int jn = 0; jn < 2; ++jn)
#pragma unroll
      for (int r = 0; r < 8; ++r) acc[jm][jn][r] = 0.0f;

  const int kiters = Kd >> 5;
  for (int kk = 0; kk < kiters; ++kk) {
    const int kg = kk << 5;
    __syncthreads();   // protect LDS from previous iteration's readers
    // ---- stage A tile: 128 rows x 32 k ----
    if constexpr (A_F32) {
#pragma unroll
      for (int e = 0; e < 16; ++e) {
        int idx = tid + e * 256;              // 0..4095
        int r = idx >> 5, kl = idx & 31;
        As[idx] = (f16)Af[(size_t)(m0 + r) * Kd + kg + kl];
      }
    } else {
      // f16 source: 16B copies (async path when available)
#pragma unroll
      for (int e = 0; e < 2; ++e) {
        int i8 = tid + e * 256;               // 0..511 (16B units)
        int r = i8 >> 2, kl = (i8 & 3) * 8;
        cp16_g2l(Ah + (size_t)(m0 + r) * Kd + kg + kl, &As[i8 * 8]);
      }
    }
    // ---- stage W tile: 32 k x 64 n, stored transposed [n][k] ----
#pragma unroll
    for (int e = 0; e < 8; ++e) {
      int idx = tid + e * 256;                // 0..2047
      int n = idx & 63, kb = idx >> 6;
      Bs[n * 32 + kb] = (f16)W[(size_t)(kg + kb) * Nd + n0 + n];
    }
    if constexpr (!A_F32) cp_wait();
    __syncthreads();

    AF a[2], b[2];
#pragma unroll
    for (int jm = 0; jm < 2; ++jm) {
      int arow = mw * 32 + jm * 16 + ma;
      a[jm].h[0] = *(const v8h*)&As[arow * 32 + k0l];
      a[jm].h[1] = *(const v8h*)&As[arow * 32 + k0l + 16];
    }
#pragma unroll
    for (int jn = 0; jn < 2; ++jn) {
      int n = nw * 32 + jn * 16 + ma;
      b[jn].h[0] = *(const v8h*)&Bs[n * 32 + k0l];
      b[jn].h[1] = *(const v8h*)&Bs[n * 32 + k0l + 16];
    }
#pragma unroll
    for (int jm = 0; jm < 2; ++jm)
#pragma unroll
      for (int jn = 0; jn < 2; ++jn)
        acc[jm][jn] = wmma16(a[jm].v, b[jn].v, acc[jm][jn]);
  }

  // epilogue: C layout rows = r + 8*(lane>=16), col = lane&15
#pragma unroll
  for (int jm = 0; jm < 2; ++jm)
#pragma unroll
    for (int jn = 0; jn < 2; ++jn) {
      int gcol = n0 + nw * 32 + jn * 16 + ma;
      float bv = bias[gcol];
#pragma unroll
      for (int r = 0; r < 8; ++r) {
        int grow = m0 + mw * 32 + jm * 16 + r + lh * 8;
        float v = acc[jm][jn][r] + bv;
        if (OUT_F32) Of[(size_t)grow * Nd + gcol] = v;
        else         Oh[(size_t)grow * Nd + gcol] = (f16)v;
      }
    }
}

// ---------------------------------------------------------------------------
// RoPE + head split: qkv f16 [M][3D] -> Q,K (rotated), V as f16 [B][H][L][HD]
// ---------------------------------------------------------------------------
__global__ __launch_bounds__(256) void rope_split(const f16* __restrict__ qkv,
                                                  f16* __restrict__ Q,
                                                  f16* __restrict__ Kh,
                                                  f16* __restrict__ V) {
  int idx = blockIdx.x * 256 + threadIdx.x;   // [0, B*L*H*HD)
  int d = idx & 63;
  int h = (idx >> 6) & 15;
  int l = (idx >> 10) & 2047;
  int b = idx >> 21;

  int j = d & 31;
  float ang = (float)l * __powf(10000.0f, -(float)j * (1.0f / 32.0f));
  float sn, cs;
  __sincosf(ang, &sn, &cs);

  size_t rowq = (size_t)(b * L + l) * (3 * D);
  int col  = h * 64 + d;
  int colp = h * 64 + (d ^ 32);

  float q  = (float)qkv[rowq + col];
  float qp = (float)qkv[rowq + colp];
  float k  = (float)qkv[rowq + D + col];
  float kp = (float)qkv[rowq + D + colp];
  float v  = (float)qkv[rowq + 2 * D + col];

  float rq = (d < 32) ? -qp : qp;   // rotate_half
  float rk = (d < 32) ? -kp : kp;

  size_t o = (size_t)((b * H + h) * L + l) * HD + d;
  Q[o]  = (f16)(q * cs + rq * sn);
  Kh[o] = (f16)(k * cs + rk * sn);
  V[o]  = (f16)v;
}

// ---------------------------------------------------------------------------
// Causal flash attention. Grid (L/64, B*H). 128 threads = 4 waves; wave w owns
// q rows [q0+16w, q0+16w+16). Streams 32-key chunks through LDS (async copy
// for Q/K tiles when available), online softmax with shfl_xor row reductions,
// P re-laid out via per-wave LDS bounce.
// ---------------------------------------------------------------------------
__global__ __launch_bounds__(128) void flash_attn(const f16* __restrict__ Q,
                                                  const f16* __restrict__ Kt,
                                                  const f16* __restrict__ V,
                                                  f16* __restrict__ Out) {
  __shared__ alignas(16) f16 Qs[64 * 64];       // [qrow][hd]
  __shared__ alignas(16) f16 Ks[32 * 64];       // [kpos][hd]
  __shared__ alignas(16) f16 Vs[64 * 32];       // transposed: [hd][kpos]
  __shared__ alignas(16) f16 Ps[4 * 16 * 32];   // per-wave P tile [m][kpos]

  const int tid  = threadIdx.x;
  const int lane = tid & 31;
  const int wv   = tid >> 5;
  const int q0   = blockIdx.x * 64;
  const int bh   = blockIdx.y;           // b*H + h
  const int h    = bh & 15;
  const int b    = bh >> 4;
  const size_t base = (size_t)bh * L * HD;

  // stage Q tile once: 64x64 f16, contiguous image of Q rows q0..q0+63
  const f16* gq = Q + base + (size_t)q0 * HD;
#pragma unroll
  for (int e = 0; e < 4; ++e) {
    int i8 = tid + e * 128;               // 0..511 (16B units)
    cp16_g2l(gq + (size_t)i8 * 8, &Qs[i8 * 8]);
  }

  const int ma   = lane & 15;
  const int lh   = lane >> 4;
  const int k0l  = lh * 8;
  const int kb16 = lh * 16;

  float mr[8], lr[8];
  v8f o[4];
#pragma unroll
  for (int r = 0; r < 8; ++r) { mr[r] = -1e30f; lr[r] = 0.0f; }
#pragma unroll
  for (int jj = 0; jj < 4; ++jj)
#pragma unroll
    for (int r = 0; r < 8; ++r) o[jj][r] = 0.0f;

  const int nch = q0 / 32 + 2;   // causal: keys up to q0+64
  for (int kc = 0; kc < nch; ++kc) {
    const int kbase = kc * 32;
    __syncthreads();             // previous chunk fully consumed
    // ---- K chunk: 32x64 f16 contiguous image ----
    const f16* gk = Kt + base + (size_t)kbase * HD;
#pragma unroll
    for (int e = 0; e < 2; ++e) {
      int i8 = tid + e * 128;             // 0..255
      cp16_g2l(gk + (size_t)i8 * 8, &Ks[i8 * 8]);
    }
    // ---- V chunk: load b128, scatter transposed [d][kpos] ----
    const f16* gv = V + base + (size_t)kbase * HD;
#pragma unroll
    for (int e = 0; e < 2; ++e) {
      int i8 = tid + e * 128;             // 0..255
      int kp = i8 >> 3, d0 = (i8 & 7) * 8;
      v8h vv = *(const v8h*)(gv + (size_t)kp * HD + d0);
#pragma unroll
      for (int t = 0; t < 8; ++t) Vs[(d0 + t) * 32 + kp] = vv[t];
    }
    cp_wait();                   // drain ASYNCcnt (Q tile on first pass too)
    __syncthreads();

    // Q A-frags for the two hd k-steps
    AF aq0, aq1;
    {
      int qrow = wv * 16 + ma;
      aq0.h[0] = *(const v8h*)&Qs[qrow * 64 + k0l];
      aq0.h[1] = *(const v8h*)&Qs[qrow * 64 + k0l + 16];
      aq1.h[0] = *(const v8h*)&Qs[qrow * 64 + 32 + k0l];
      aq1.h[1] = *(const v8h*)&Qs[qrow * 64 + 32 + k0l + 16];
    }

    // S = Q * K^T  (two 16x16 column tiles over the 32-key chunk)
    v8f s[2];
#pragma unroll
    for (int j = 0; j < 2; ++j) {
      int n = j * 16 + ma;                // key position in chunk
      AF b0, b1;
      b0.h[0] = *(const v8h*)&Ks[n * 64 + k0l];
      b0.h[1] = *(const v8h*)&Ks[n * 64 + k0l + 16];
      b1.h[0] = *(const v8h*)&Ks[n * 64 + 32 + k0l];
      b1.h[1] = *(const v8h*)&Ks[n * 64 + 32 + k0l + 16];
      v8f z;
#pragma unroll
      for (int r = 0; r < 8; ++r) z[r] = 0.0f;
      z = wmma16(aq0.v, b0.v, z);
      z = wmma16(aq1.v, b1.v, z);
      s[j] = z;
    }

    // online softmax per row (row m = r + 8*(lane>=16); 16 lanes share a row)
    float alpha[8];
    const int wbase = wv * 16 * 32;
#pragma unroll
    for (int r = 0; r < 8; ++r) {
      int mrow = r + lh * 8;
      int qpos = q0 + wv * 16 + mrow;
      float v0 = s[0][r] * SCALE;
      float v1 = s[1][r] * SCALE;
      if (kbase + ma      > qpos) v0 = -1e30f;  // causal mask
      if (kbase + 16 + ma > qpos) v1 = -1e30f;
      float rm = fmaxf(v0, v1);
      for (int off = 1; off < 16; off <<= 1)
        rm = fmaxf(rm, __shfl_xor(rm, off, 32));
      float mnew = fmaxf(mr[r], rm);
      float a  = __expf(mr[r] - mnew);
      float p0 = __expf(v0 - mnew);
      float p1 = __expf(v1 - mnew);
      float ps = p0 + p1;
      for (int off = 1; off < 16; off <<= 1)
        ps += __shfl_xor(ps, off, 32);
      lr[r] = lr[r] * a + ps;
      mr[r] = mnew;
      alpha[r] = a;
      Ps[wbase + mrow * 32 + ma]      = (f16)p0;
      Ps[wbase + mrow * 32 + 16 + ma] = (f16)p1;
    }
#pragma unroll
    for (int jj = 0; jj < 4; ++jj)
#pragma unroll
      for (int r = 0; r < 8; ++r) o[jj][r] *= alpha[r];

    // gather P as A-frag (same-wave LDS bounce; DScnt wait inserted by compiler)
    AF ap;
    ap.h[0] = *(const v8h*)&Ps[wbase + ma * 32 + k0l];
    ap.h[1] = *(const v8h*)&Ps[wbase + ma * 32 + k0l + 16];

    // O += P * V  (4 hd column tiles)
#pragma unroll
    for (int jj = 0; jj < 4; ++jj) {
      int n = jj * 16 + ma;               // hd column
      AF bv;
      bv.h[0] = *(const v8h*)&Vs[n * 32 + k0l];
      bv.h[1] = *(const v8h*)&Vs[n * 32 + k0l + 16];
      o[jj] = wmma16(ap.v, bv.v, o[jj]);
    }
  }

  // normalize and write [M][D] f16, col = h*64 + d
#pragma unroll
  for (int jj = 0; jj < 4; ++jj)
#pragma unroll
    for (int r = 0; r < 8; ++r) {
      int lrow = q0 + wv * 16 + r + lh * 8;
      int col  = h * 64 + jj * 16 + ma;
      Out[(size_t)(b * L + lrow) * D + col] = (f16)(o[jj][r] / lr[r]);
    }
}

// ---------------------------------------------------------------------------
extern "C" void kernel_launch(void* const* d_in, const int* in_sizes, int n_in,
                              void* d_out, int out_size, void* d_ws, size_t ws_size,
                              hipStream_t stream) {
  (void)in_sizes; (void)n_in; (void)out_size; (void)ws_size;
  const float* x      = (const float*)d_in[0];
  const float* qkv_w  = (const float*)d_in[1];
  const float* qkv_b  = (const float*)d_in[2];
  const float* proj_w = (const float*)d_in[3];
  const float* proj_b = (const float*)d_in[4];

  char* ws   = (char*)d_ws;
  f16*  qkvh = (f16*)(ws + OFF_QKV);
  f16*  Qh   = (f16*)(ws + OFF_Q);
  f16*  Kh   = (f16*)(ws + OFF_K);
  f16*  Vh   = (f16*)(ws + OFF_V);
  f16*  At   = (f16*)(ws + OFF_AT);
  float* out = (float*)d_out;

  // 1) qkv = x @ qkv_w + qkv_b   (f32 in, f16 out)
  gemm_tile128<true, false><<<dim3(M / 128, (3 * D) / 64), 256, 0, stream>>>(
      x, qkv_w, qkv_b, qkvh, D, 3 * D);
  // 2) RoPE + head split
  rope_split<<<(Bsz * L * H * HD) / 256, 256, 0, stream>>>(qkvh, Qh, Kh, Vh);
  // 3) causal flash attention
  flash_attn<<<dim3(L / 64, Bsz * H), 128, 0, stream>>>(Qh, Kh, Vh, At);
  // 4) out = attn @ proj_w + proj_b   (f16 in, f32 out)
  gemm_tile128<false, true><<<dim3(M / 128, D / 64), 256, 0, stream>>>(
      At, proj_w, proj_b, out, D, D);
}
